// CodeLACE_59133109731394
// MI455X (gfx1250) — matrix operand order repaired and checked by reading, not compiled, not run on hardware
//
#include <hip/hip_runtime.h>
#include <hip/hip_bf16.h>

// ---------------- problem constants (match reference) ----------------
#define HDIM   1024
#define NHEAD  16
#define HDSZ   64
#define NEXP   4
#define IDIM   4096
#define NLAY   2
#define BBAT   2
#define SSEQ   1024
#define NCS    10
#define NCM    20
#define NCP    15
#define LNEPS  1e-12f

// ---------------- WMMA types ----------------
typedef __bf16 bf16_t;
typedef bf16_t v16bf __attribute__((ext_vector_type(16)));
typedef bf16_t v8bf  __attribute__((ext_vector_type(8)));
typedef float  v8f   __attribute__((ext_vector_type(8)));

union Frag16 { v16bf v; v8bf h[2]; unsigned short u[16]; };

__device__ __forceinline__ unsigned short f2bf(float f) {
    unsigned int u = __float_as_uint(f);
    u += 0x7FFFu + ((u >> 16) & 1u);   // round-to-nearest-even
    return (unsigned short)(u >> 16);
}
__device__ __forceinline__ float bf2f(unsigned short u) {
    return __uint_as_float((unsigned int)u << 16);
}

// ---------------- generic bf16 WMMA GEMM (bf16 operands in memory) ----------------
// C[M,N] = epilogue( alpha * A[M,K] x B[K,N] + bias[N] )
// transB: B element (k,n) at n*ldb+k (for Q@K^T)
// epilogue: relu; rowScale[m*rsStride]; accumulate (fp32 C only); out_bf16
// batching: z1 = z >> batchShift, z2 = z & (2^batchShift - 1)
struct GemmP {
    const unsigned short* A; const unsigned short* B; void* C;
    const float* bias; const float* rowScale;
    int M, N, K;
    int lda, ldb, ldc;
    int transB, relu, accum, out_bf16, rsStride, batchShift;
    float alpha;
    long long sA1, sA2, sB1, sB2, sC1, sC2;
};

#define BM  64
#define BN  128
#define BK  32
#define BKP 40   // padded row (80B = 20 banks) -> conflict-free fragment reads

template<bool CHECK>
__global__ __launch_bounds__(256) void gemm_bf16_wmma(GemmP p) {
    // double-buffered tiles: 2*(64+128)*40*2B = 30 KB
    __shared__ __align__(16) unsigned short As[2][BM][BKP];   // [m][k]
    __shared__ __align__(16) unsigned short Bs[2][BN][BKP];   // [n][k]

    const int t    = threadIdx.x;
    const int wave = t >> 5, lane = t & 31;
    const int lr   = lane & 15, hlf = lane >> 4;
    const int wm   = wave & 1, wn = wave >> 1;                // 2 x 4 wave grid

    const int z  = blockIdx.z;
    const int z1 = z >> p.batchShift;
    const int z2 = z & ((1 << p.batchShift) - 1);
    const unsigned short* A = p.A + (long long)z1 * p.sA1 + (long long)z2 * p.sA2;
    const unsigned short* B = p.B + (long long)z1 * p.sB1 + (long long)z2 * p.sB2;
    float*          Cf = (float*)p.C          + (long long)z1 * p.sC1 + (long long)z2 * p.sC2;
    unsigned short* Ch = (unsigned short*)p.C + (long long)z1 * p.sC1 + (long long)z2 * p.sC2;

    const int m0 = blockIdx.y * BM;
    const int n0 = blockIdx.x * BN;

    // stage one K-tile (global bf16 -> LDS bf16) into buffer `buf`
    auto stage = [&](int buf, int k0) {
        if (!CHECK) {
            // ---- A: one uint4 (8 bf16) per thread ----
            {
                int r = t >> 2, c8 = t & 3;
                uint4 av = *(const uint4*)&A[(long long)(m0 + r) * p.lda + k0 + c8 * 8];
                *(uint4*)&As[buf][r][c8 * 8] = av;
            }
            if (p.transB) {
                // contiguous along k: 2 x uint4 per thread
                #pragma unroll
                for (int it = 0; it < 2; ++it) {
                    int j = t + it * 256;      // 0..511
                    int n = j >> 2, c8 = j & 3;
                    uint4 bv = *(const uint4*)&B[(long long)(n0 + n) * p.ldb + k0 + c8 * 8];
                    *(uint4*)&Bs[buf][n][c8 * 8] = bv;
                }
            } else {
                // contiguous along n; scatter 8 bf16 to 8 LDS rows
                #pragma unroll
                for (int it = 0; it < 2; ++it) {
                    int j = t + it * 256;      // 0..511
                    int n8 = (j & 15) * 8, c = j >> 4;     // c = 0..31
                    uint4 bv = *(const uint4*)&B[(long long)(k0 + c) * p.ldb + n0 + n8];
                    Bs[buf][n8 + 0][c] = (unsigned short)(bv.x & 0xFFFF);
                    Bs[buf][n8 + 1][c] = (unsigned short)(bv.x >> 16);
                    Bs[buf][n8 + 2][c] = (unsigned short)(bv.y & 0xFFFF);
                    Bs[buf][n8 + 3][c] = (unsigned short)(bv.y >> 16);
                    Bs[buf][n8 + 4][c] = (unsigned short)(bv.z & 0xFFFF);
                    Bs[buf][n8 + 5][c] = (unsigned short)(bv.z >> 16);
                    Bs[buf][n8 + 6][c] = (unsigned short)(bv.w & 0xFFFF);
                    Bs[buf][n8 + 7][c] = (unsigned short)(bv.w >> 16);
                }
            }
            // speculative prefetch of the tile after next (global_prefetch_b8)
            if (k0 + 2 * BK < p.K)
                __builtin_prefetch(&A[(long long)(m0 + (t >> 2)) * p.lda + k0 + 2 * BK], 0, 1);
        } else {
            // ---- branchless guarded path (safe-address select) ----
            #pragma unroll
            for (int i = t; i < BM * BK; i += 256) {
                int r = i >> 5, c = i & 31;
                int gr = m0 + r, gc = k0 + c;
                bool ok = (gr < p.M) && (gc < p.K);
                long long off = ok ? ((long long)gr * p.lda + gc) : 0;
                unsigned short v = A[off];
                As[buf][r][c] = ok ? v : (unsigned short)0;
            }
            if (p.transB) {
                #pragma unroll
                for (int i = t; i < BN * BK; i += 256) {
                    int n = i >> 5, c = i & 31;
                    int gn = n0 + n, gc = k0 + c;
                    bool ok = (gn < p.N) && (gc < p.K);
                    long long off = ok ? ((long long)gn * p.ldb + gc) : 0;
                    unsigned short v = B[off];
                    Bs[buf][n][c] = ok ? v : (unsigned short)0;
                }
            } else {
                #pragma unroll
                for (int i = t; i < BN * BK; i += 256) {
                    int n = i & (BN - 1), c = i >> 7;
                    int gn = n0 + n, gc = k0 + c;
                    bool ok = (gn < p.N) && (gc < p.K);
                    long long off = ok ? ((long long)gc * p.ldb + gn) : 0;
                    unsigned short v = B[off];
                    Bs[buf][n][c] = ok ? v : (unsigned short)0;
                }
            }
        }
    };

    v8f acc[2][2];
    {
        v8f vz;
        #pragma unroll
        for (int r = 0; r < 8; ++r) vz[r] = 0.f;
        acc[0][0] = vz; acc[0][1] = vz; acc[1][0] = vz; acc[1][1] = vz;
    }

    const int nk = p.K / BK;   // all call sites have K % 32 == 0
    stage(0, 0);
    __syncthreads();

    for (int kt = 0; kt < nk; ++kt) {
        const int cur = kt & 1;

        // ---- fragments: contiguous k order (k = 16*hlf + s) for A and B ----
        Frag16 fa[2], fb[2];
        #pragma unroll
        for (int mg = 0; mg < 2; ++mg) {
            int row = wm * 32 + mg * 16 + lr;
            // A 16x32: lanes 0-15 K=0..15; lanes 16-31 K=16..31 (matches B)
            fa[mg].h[0] = *(const v8bf*)&As[cur][row][hlf * 16];
            fa[mg].h[1] = *(const v8bf*)&As[cur][row][hlf * 16 + 8];
        }
        #pragma unroll
        for (int ng = 0; ng < 2; ++ng) {
            int col = wn * 32 + ng * 16 + lr;
            // B 32x16: lane holds one column; contiguous K half per lane group
            fb[ng].v = *(const v16bf*)&Bs[cur][col][hlf * 16];
        }

        // kick off next tile's global->LDS while WMMAs run
        if (kt + 1 < nk) stage(cur ^ 1, (kt + 1) * BK);

        #pragma unroll
        for (int mg = 0; mg < 2; ++mg)
            #pragma unroll
            for (int ng = 0; ng < 2; ++ng)
                acc[mg][ng] = __builtin_amdgcn_wmma_f32_16x16x32_bf16(
                    false, fa[mg].v, false, fb[ng].v,
                    (short)0, acc[mg][ng], false, false);

        __syncthreads();
    }

    // ---- epilogue: C/D layout: lane0-15 N=lane M=r; lane16-31 N=lane-16 M=r+8 ----
    #pragma unroll
    for (int mg = 0; mg < 2; ++mg) {
        #pragma unroll
        for (int ng = 0; ng < 2; ++ng) {
            int cn = n0 + wn * 32 + ng * 16 + lr;
            if (CHECK && cn >= p.N) continue;
            int cmBase = m0 + wm * 32 + mg * 16 + hlf * 8;
            float bv = p.bias ? p.bias[cn] : 0.f;
            #pragma unroll
            for (int r = 0; r < 8; ++r) {
                int cm = cmBase + r;
                if (CHECK && cm >= p.M) break;
                float vres = acc[mg][ng][r] * p.alpha + bv;
                if (p.relu) vres = fmaxf(vres, 0.f);
                if (p.rowScale) vres *= p.rowScale[(long long)cm * p.rsStride];
                long long off = (long long)cm * p.ldc + cn;
                if (p.out_bf16)     Ch[off] = f2bf(vres);
                else if (p.accum)   Cf[off] += vres;
                else                Cf[off] = vres;
            }
        }
    }
}

// ---------------- fp32 -> bf16 bulk convert (weights / activations) ----------------
__global__ void cvt_bf16_kernel(const float* __restrict__ src, unsigned short* __restrict__ dst,
                                long long n4) {
    long long i = (long long)blockIdx.x * blockDim.x + threadIdx.x;
    if (i >= n4) return;
    float4 v = ((const float4*)src)[i];
    uint2 o{(unsigned)f2bf(v.x) | ((unsigned)f2bf(v.y) << 16),
            (unsigned)f2bf(v.z) | ((unsigned)f2bf(v.w) << 16)};
    ((uint2*)dst)[i] = o;
}

// ---------------- TDM prefetch (CDNA5 Tensor Data Mover) ----------------
#if defined(__has_builtin)
#if __has_builtin(__builtin_amdgcn_tensor_load_to_lds)
#define HAVE_TDM 1
#endif
#endif

#ifdef HAVE_TDM
typedef unsigned int u32x4 __attribute__((ext_vector_type(4)));
typedef int          i32x4 __attribute__((ext_vector_type(4)));
typedef int          i32x8 __attribute__((ext_vector_type(8)));

// Pull the first 32x128 bf16 tile of a [1024][1024] bf16 weight into LDS via TDM.
// amdgpu-toolchain (clang-23) builtin: 6 args (g0 u32x4, g1 i32x8, g2 i32x4, g3 i32x4,
// g4 i32x8, i32 cpol).
__global__ void tdm_prefetch_kernel(const unsigned short* __restrict__ w,
                                    float* __restrict__ sink) {
    __shared__ __align__(16) unsigned short buf[32 * 128];
    unsigned long long ga = (unsigned long long)(const void*)w;
    u32x4 g0;
    g0[0] = 1u;                                            // count=1, user mode
    g0[1] = 0u;                                            // lds_addr = 0
    g0[2] = (unsigned)(ga & 0xFFFFFFFFu);                  // global_addr[31:0]
    g0[3] = (unsigned)((ga >> 32) & 0x1FFFFFFu) | (2u << 30);  // addr[56:32] | type=2
    i32x8 g1;
    g1[0] = 0x00010000;                                    // wg_mask=0, data_size=1 (2B)
    g1[1] = (int)((1024u & 0xFFFFu) << 16);                // tensor_dim0[15:0]
    g1[2] = (int)((1024u >> 16) | ((1024u & 0xFFFFu) << 16)); // dim0 hi | dim1 lo
    g1[3] = (int)((1024u >> 16) | (128u << 16));           // dim1 hi | tile_dim0=128
    g1[4] = 32;                                            // tile_dim1=32, tile_dim2=0
    g1[5] = 1024;                                          // tensor_dim0_stride lo
    g1[6] = 0;
    g1[7] = 0;
    i32x4 g2; g2[0] = 0; g2[1] = 0; g2[2] = 0; g2[3] = 0;  // 2D: groups 2/3 unused
    i32x4 g3 = g2;
    i32x8 g4;                                              // extra group (clang-23 form)
    g4[0] = 0; g4[1] = 0; g4[2] = 0; g4[3] = 0;
    g4[4] = 0; g4[5] = 0; g4[6] = 0; g4[7] = 0;
    __builtin_amdgcn_tensor_load_to_lds(g0, g1, g2, g3, g4, 0);
#if __has_builtin(__builtin_amdgcn_s_wait_tensorcnt)
    __builtin_amdgcn_s_wait_tensorcnt(0);
#endif
    sink[threadIdx.x] = bf2f(buf[threadIdx.x]);            // consume (keeps LDS alive)
}
#endif

// ---------------- embedding lookup ----------------
__global__ void embed_kernel(const int* __restrict__ ids, const float* __restrict__ tok,
                             const float* __restrict__ pos, float* __restrict__ x) {
    int row = blockIdx.x;                       // b*SSEQ + s
    int s = row % SSEQ;
    long long id = ids[row];
    for (int i = threadIdx.x; i < HDIM; i += blockDim.x)
        x[(long long)row * HDIM + i] = tok[id * HDIM + i] + pos[(long long)s * HDIM + i];
}

// ---------------- (residual +) LayerNorm; fp32 out + optional bf16 mirror ----------------
__global__ __launch_bounds__(256) void ln_kernel(const float* __restrict__ x,
                                                 const float* __restrict__ res,
                                                 float* __restrict__ y,
                                                 unsigned short* __restrict__ yb,
                                                 const float* __restrict__ w,
                                                 const float* __restrict__ b) {
    __shared__ float red[256];
    long long row = blockIdx.x;
    const float* xr = x + row * HDIM;
    const float* rr = res ? res + row * HDIM : nullptr;
    float vloc[4]; float s = 0.f;
    #pragma unroll
    for (int i = 0; i < 4; ++i) {
        int h = threadIdx.x + i * 256;
        float v = xr[h] + (rr ? rr[h] : 0.f);
        vloc[i] = v; s += v;
    }
    red[threadIdx.x] = s; __syncthreads();
    for (int st = 128; st > 0; st >>= 1) {
        if (threadIdx.x < st) red[threadIdx.x] += red[threadIdx.x + st];
        __syncthreads();
    }
    float mu = red[0] / HDIM; __syncthreads();
    float vs = 0.f;
    #pragma unroll
    for (int i = 0; i < 4; ++i) { float d = vloc[i] - mu; vs += d * d; }
    red[threadIdx.x] = vs; __syncthreads();
    for (int st = 128; st > 0; st >>= 1) {
        if (threadIdx.x < st) red[threadIdx.x] += red[threadIdx.x + st];
        __syncthreads();
    }
    float inv = rsqrtf(red[0] / HDIM + LNEPS);
    #pragma unroll
    for (int i = 0; i < 4; ++i) {
        int h = threadIdx.x + i * 256;
        float o = (vloc[i] - mu) * inv * w[h] + b[h];
        y[row * HDIM + h] = o;
        if (yb) yb[row * HDIM + h] = f2bf(o);
    }
}

// ---------------- sparse mask + softmax (bf16 in/out, in place) ----------------
__global__ __launch_bounds__(256) void softmax_mask_kernel(unsigned short* __restrict__ probs,
                                                           int layer) {
    __shared__ float red[256];
    long long row = blockIdx.x;                 // (b*NHEAD+h)*SSEQ + q
    int q = (int)(row % SSEQ);
    unsigned short* p = probs + row * SSEQ;
    float vals[4]; float mx = -1e30f;
    #pragma unroll
    for (int i = 0; i < 4; ++i) {
        int j = threadIdx.x + i * 256;
        float v = bf2f(p[j]);
        int d = q - j; if (d < 0) d = -d;
        int local = (d <= 2);
        // deterministic pseudo-random ~10% columns (jax PRNG not reproducible in-kernel)
        unsigned hsh = (unsigned)q * 2654435761u ^ (unsigned)j * 40503u ^ (unsigned)layer * 2246822519u;
        hsh ^= hsh >> 13; hsh *= 1664525u; hsh ^= hsh >> 16;
        int rnd = (hsh & 1023u) < 102u;
        v *= (local || rnd) ? 1.f : 0.f;        // multiplicative mask, as in reference
        vals[i] = v; mx = fmaxf(mx, v);
    }
    red[threadIdx.x] = mx; __syncthreads();
    for (int st = 128; st > 0; st >>= 1) {
        if (threadIdx.x < st) red[threadIdx.x] = fmaxf(red[threadIdx.x], red[threadIdx.x + st]);
        __syncthreads();
    }
    mx = red[0]; __syncthreads();
    float s = 0.f;
    #pragma unroll
    for (int i = 0; i < 4; ++i) { vals[i] = __expf(vals[i] - mx); s += vals[i]; }
    red[threadIdx.x] = s; __syncthreads();
    for (int st = 128; st > 0; st >>= 1) {
        if (threadIdx.x < st) red[threadIdx.x] += red[threadIdx.x + st];
        __syncthreads();
    }
    float inv = 1.f / red[0];
    #pragma unroll
    for (int i = 0; i < 4; ++i) p[threadIdx.x + i * 256] = f2bf(vals[i] * inv);
}

// ---------------- MoE gate softmax, block per token ----------------
__global__ __launch_bounds__(128) void gate_kernel(const float* __restrict__ x,
                                                   const float* __restrict__ gw,
                                                   const float* __restrict__ gb,
                                                   float* __restrict__ gates) {
    __shared__ float red[NEXP][128];
    long long row = blockIdx.x;
    float acc[NEXP] = {0.f, 0.f, 0.f, 0.f};
    for (int h = threadIdx.x; h < HDIM; h += 128) {
        float xv = x[row * HDIM + h];
        #pragma unroll
        for (int e = 0; e < NEXP; ++e) acc[e] += xv * gw[(long long)h * NEXP + e];
    }
    #pragma unroll
    for (int e = 0; e < NEXP; ++e) red[e][threadIdx.x] = acc[e];
    __syncthreads();
    for (int st = 64; st > 0; st >>= 1) {
        if (threadIdx.x < st)
            #pragma unroll
            for (int e = 0; e < NEXP; ++e) red[e][threadIdx.x] += red[e][threadIdx.x + st];
        __syncthreads();
    }
    if (threadIdx.x == 0) {
        float g[NEXP], mx = -1e30f, s = 0.f;
        #pragma unroll
        for (int e = 0; e < NEXP; ++e) { g[e] = red[e][0] + gb[e]; mx = fmaxf(mx, g[e]); }
        #pragma unroll
        for (int e = 0; e < NEXP; ++e) { g[e] = __expf(g[e] - mx); s += g[e]; }
        #pragma unroll
        for (int e = 0; e < NEXP; ++e) gates[row * NEXP + e] = g[e] / s;
    }
}

__global__ void zero_kernel(float* __restrict__ p, long long n) {
    long long i = (long long)blockIdx.x * blockDim.x + threadIdx.x;
    if (i < n) p[i] = 0.f;
}

// ---------------- mean pool over sequence ----------------
__global__ void pool_kernel(const float* __restrict__ x, float* __restrict__ pooled) {
    int i = blockIdx.x * blockDim.x + threadIdx.x;      // b*HDIM + h
    if (i >= BBAT * HDIM) return;
    int b = i / HDIM, h = i % HDIM;
    float s = 0.f;
    for (int t = 0; t < SSEQ; ++t) s += x[((long long)b * SSEQ + t) * HDIM + h];
    pooled[i] = s * (1.f / SSEQ);
}

// ---------------- classifier heads: block per output scalar ----------------
__global__ __launch_bounds__(256) void head_kernel(const float* __restrict__ pooled,
        const float* __restrict__ syn_w, const float* __restrict__ syn_b,
        const float* __restrict__ sem_w, const float* __restrict__ sem_b,
        const float* __restrict__ prag_w, const float* __restrict__ prag_b,
        float* __restrict__ out) {
    __shared__ float red[256];
    int o = blockIdx.x;     // layout: syn[B][CS] | sem[B][CM] | prag[B][CP]
    const float *w, *bb; int b, c, nc;
    if (o < BBAT * NCS) {
        b = o / NCS; c = o % NCS; w = syn_w; bb = syn_b; nc = NCS;
    } else if (o < BBAT * NCS + BBAT * NCM) {
        int oo = o - BBAT * NCS; b = oo / NCM; c = oo % NCM; w = sem_w; bb = sem_b; nc = NCM;
    } else {
        int oo = o - BBAT * NCS - BBAT * NCM; b = oo / NCP; c = oo % NCP; w = prag_w; bb = prag_b; nc = NCP;
    }
    float s = 0.f;
    for (int h = threadIdx.x; h < HDIM; h += 256)
        s += pooled[(long long)b * HDIM + h] * w[(long long)h * nc + c];
    red[threadIdx.x] = s; __syncthreads();
    for (int st = 128; st > 0; st >>= 1) {
        if (threadIdx.x < st) red[threadIdx.x] += red[threadIdx.x + st];
        __syncthreads();
    }
    if (threadIdx.x == 0) out[o] = red[0] + bb[c];
}

// ---------------- host-side GEMM launcher ----------------
static void launch_gemm(hipStream_t stream, const unsigned short* A, const unsigned short* B,
                        void* C, const float* bias, int M, int N, int K,
                        int lda, int ldb, int ldc, int transB, float alpha,
                        int relu, int accum, int out_bf16,
                        const float* rowScale, int rsStride,
                        int batch, int batchShift,
                        long long sA1, long long sA2, long long sB1, long long sB2,
                        long long sC1, long long sC2) {
    GemmP p;
    p.A = A; p.B = B; p.C = C; p.bias = bias; p.rowScale = rowScale;
    p.M = M; p.N = N; p.K = K; p.lda = lda; p.ldb = ldb; p.ldc = ldc;
    p.transB = transB; p.relu = relu; p.accum = accum; p.out_bf16 = out_bf16;
    p.rsStride = rsStride; p.batchShift = batchShift; p.alpha = alpha;
    p.sA1 = sA1; p.sA2 = sA2; p.sB1 = sB1; p.sB2 = sB2; p.sC1 = sC1; p.sC2 = sC2;
    dim3 grid((N + BN - 1) / BN, (M + BM - 1) / BM, batch);
    bool check = (M % BM) || (N % BN) || (K % BK);
    if (check) gemm_bf16_wmma<true ><<<grid, 256, 0, stream>>>(p);
    else       gemm_bf16_wmma<false><<<grid, 256, 0, stream>>>(p);
}

static void launch_cvt(hipStream_t stream, const float* src, unsigned short* dst, long long n) {
    long long n4 = n / 4;
    cvt_bf16_kernel<<<(int)((n4 + 255) / 256), 256, 0, stream>>>(src, dst, n4);
}

extern "C" void kernel_launch(void* const* d_in, const int* in_sizes, int n_in,
                              void* d_out, int out_size, void* d_ws, size_t ws_size,
                              hipStream_t stream) {
    (void)in_sizes; (void)n_in; (void)out_size; (void)ws_size;
    // inputs in setup_inputs() order
    const int*   input_ids = (const int*)  d_in[0];
    const float* tok_emb   = (const float*)d_in[1];
    const float* pos_emb   = (const float*)d_in[2];
    const float* ln0_w     = (const float*)d_in[3];
    const float* ln0_b     = (const float*)d_in[4];
    const float* Wq        = (const float*)d_in[5];
    const float* bq        = (const float*)d_in[6];
    const float* Wk        = (const float*)d_in[7];
    const float* bk        = (const float*)d_in[8];
    const float* Wv        = (const float*)d_in[9];
    const float* bv        = (const float*)d_in[10];
    const float* ln1_w     = (const float*)d_in[11];
    const float* ln1_b     = (const float*)d_in[12];
    const float* gate_w    = (const float*)d_in[13];
    const float* gate_b    = (const float*)d_in[14];
    const float* ew1       = (const float*)d_in[15];
    const float* eb1       = (const float*)d_in[16];
    const float* ew2       = (const float*)d_in[17];
    const float* eb2       = (const float*)d_in[18];
    const float* ln2_w     = (const float*)d_in[19];
    const float* ln2_b     = (const float*)d_in[20];
    const float* syn_w     = (const float*)d_in[21];
    const float* syn_b     = (const float*)d_in[22];
    const float* sem_w     = (const float*)d_in[23];
    const float* sem_b     = (const float*)d_in[24];
    const float* prag_w    = (const float*)d_in[25];
    const float* prag_b    = (const float*)d_in[26];

    const long long NTOK = (long long)BBAT * SSEQ;            // 2048
    const long long WQKV = (long long)NLAY * HDIM * HDIM;     // per-matrix weight elems
    const long long WEXP = (long long)NLAY * NEXP * HDIM * IDIM;

    // ---- workspace layout: fp32 region then bf16 region (~270 MB total) ----
    float* x0     = (float*)d_ws;
    float* x1     = x0 + NTOK * HDIM;
    float* ctx    = x1 + NTOK * HDIM;
    float* moe    = ctx + NTOK * HDIM;
    float* gates  = moe + NTOK * HDIM;
    float* pooled = gates + NTOK * NEXP;
    float* sink   = pooled + BBAT * HDIM;
    unsigned short* xb     = (unsigned short*)(sink + 64);
    unsigned short* qb     = xb + NTOK * HDIM;
    unsigned short* kb     = qb + NTOK * HDIM;
    unsigned short* vb     = kb + NTOK * HDIM;
    unsigned short* h1b    = vb + NTOK * HDIM;
    unsigned short* probsb = h1b + NTOK * IDIM;
    unsigned short* wqb    = probsb + (long long)BBAT * NHEAD * SSEQ * SSEQ;
    unsigned short* wkb    = wqb + WQKV;
    unsigned short* wvb    = wkb + WQKV;
    unsigned short* ew1b   = wvb + WQKV;
    unsigned short* ew2b   = ew1b + WEXP;

    // ---- one-time (per launch) weight conversion fp32 -> bf16 ----
    launch_cvt(stream, Wq,  wqb,  WQKV);
    launch_cvt(stream, Wk,  wkb,  WQKV);
    launch_cvt(stream, Wv,  wvb,  WQKV);
    launch_cvt(stream, ew1, ew1b, WEXP);
    launch_cvt(stream, ew2, ew2b, WEXP);

#ifdef HAVE_TDM
    // exercise the Tensor Data Mover: DMA the first weight tile into LDS
    tdm_prefetch_kernel<<<1, 32, 0, stream>>>(wqb, sink);
#endif

    // ---- embedding + LN0 (fp32 x + bf16 mirror) ----
    embed_kernel<<<(int)NTOK, 256, 0, stream>>>(input_ids, tok_emb, pos_emb, x0);
    ln_kernel<<<(int)NTOK, 256, 0, stream>>>(x0, nullptr, x1, xb, ln0_w, ln0_b);
    float* x = x1; float* xalt = x0;

    for (int l = 0; l < NLAY; ++l) {
        const long long wOff = (long long)l * HDIM * HDIM;
        // QKV projections (bf16 in, bf16 out)
        launch_gemm(stream, xb, wqb + wOff, qb, bq + l * HDIM, (int)NTOK, HDIM, HDIM,
                    HDIM, HDIM, HDIM, 0, 1.f, 0, 0, 1, nullptr, 0, 1, 0, 0, 0, 0, 0, 0, 0);
        launch_gemm(stream, xb, wkb + wOff, kb, bk + l * HDIM, (int)NTOK, HDIM, HDIM,
                    HDIM, HDIM, HDIM, 0, 1.f, 0, 0, 1, nullptr, 0, 1, 0, 0, 0, 0, 0, 0, 0);
        launch_gemm(stream, xb, wvb + wOff, vb, bv + l * HDIM, (int)NTOK, HDIM, HDIM,
                    HDIM, HDIM, HDIM, 0, 1.f, 0, 0, 1, nullptr, 0, 1, 0, 0, 0, 0, 0, 0, 0);
        // scores = Q K^T / sqrt(HD), batched over (b,h); z = b*NHEAD + h
        launch_gemm(stream, qb, kb, probsb, nullptr, SSEQ, SSEQ, HDSZ,
                    HDIM, HDIM, SSEQ, /*transB*/1, 0.125f, 0, 0, 1, nullptr, 0,
                    BBAT * NHEAD, 4,
                    (long long)SSEQ * HDIM, HDSZ,
                    (long long)SSEQ * HDIM, HDSZ,
                    (long long)NHEAD * SSEQ * SSEQ, (long long)SSEQ * SSEQ);
        softmax_mask_kernel<<<BBAT * NHEAD * SSEQ, 256, 0, stream>>>(probsb, l);
        // ctx = probs @ V, batched (N=64 -> guarded kernel), fp32 out for residual
        launch_gemm(stream, probsb, vb, ctx, nullptr, SSEQ, HDSZ, SSEQ,
                    SSEQ, HDIM, HDIM, 0, 1.f, 0, 0, 0, nullptr, 0,
                    BBAT * NHEAD, 4,
                    (long long)NHEAD * SSEQ * SSEQ, (long long)SSEQ * SSEQ,
                    (long long)SSEQ * HDIM, HDSZ,
                    (long long)SSEQ * HDIM, HDSZ);
        // x = LN(x + ctx)  (fp32 + bf16 mirror)
        ln_kernel<<<(int)NTOK, 256, 0, stream>>>(x, ctx, xalt, xb,
                                                 ln1_w + l * HDIM, ln1_b + l * HDIM);
        { float* t0 = x; x = xalt; xalt = t0; }
        // MoE
        gate_kernel<<<(int)NTOK, 128, 0, stream>>>(x, gate_w + (long long)l * HDIM * NEXP,
                                                   gate_b + l * NEXP, gates);
        zero_kernel<<<(int)((NTOK * HDIM + 255) / 256), 256, 0, stream>>>(moe, NTOK * HDIM);
        for (int e = 0; e < NEXP; ++e) {
            const long long ei = (long long)(l * NEXP + e);
            // h1 = relu(x @ w1 + b1) -> bf16
            launch_gemm(stream, xb, ew1b + ei * HDIM * IDIM, h1b, eb1 + ei * IDIM,
                        (int)NTOK, IDIM, HDIM, HDIM, IDIM, IDIM, 0, 1.f,
                        1, 0, 1, nullptr, 0, 1, 0, 0, 0, 0, 0, 0, 0);
            // moe += gates[:,e] * (h1 @ w2 + b2)  (fp32 accumulate)
            launch_gemm(stream, h1b, ew2b + ei * IDIM * HDIM, moe, eb2 + ei * HDIM,
                        (int)NTOK, HDIM, IDIM, IDIM, HDIM, HDIM, 0, 1.f,
                        0, 1, 0, gates + e, NEXP, 1, 0, 0, 0, 0, 0, 0, 0);
        }
        // x = LN(x + moe)  (fp32 + bf16 mirror)
        ln_kernel<<<(int)NTOK, 256, 0, stream>>>(x, moe, xalt, xb,
                                                 ln2_w + l * HDIM, ln2_b + l * HDIM);
        { float* t0 = x; x = xalt; xalt = t0; }
    }

    pool_kernel<<<(BBAT * HDIM + 255) / 256, 256, 0, stream>>>(x, pooled);
    head_kernel<<<BBAT * (NCS + NCM + NCP), 256, 0, stream>>>(
        pooled, syn_w, syn_b, sem_w, sem_b, prag_w, prag_b, (float*)d_out);
}
